// Discriminator_65695819760469
// MI455X (gfx1250) — matrix operand order repaired and checked by reading, not compiled, not run on hardware
//
#include <hip/hip_runtime.h>
#include <hip/hip_bf16.h>
#include <math.h>

#define IN_DIM   500
#define BATCH    131072
#define KPAD     512
#define KSTRIDE  520          // padded LDS row stride (halves) -> conflict-free b128 reads
#define NBSECTOR 11
#define NBMQ     10

typedef __bf16 v16bf __attribute__((ext_vector_type(16)));
typedef float  v8f   __attribute__((ext_vector_type(8)));

union BF16Vec { v16bf v; unsigned short u[16]; uint4 q[2]; };

__device__ __forceinline__ unsigned short f32_to_bf16_rn(float f) {
  unsigned u = __float_as_uint(f);
  u += 0x7FFFu + ((u >> 16) & 1u);           // round-to-nearest-even
  return (unsigned short)(u >> 16);
}
__device__ __forceinline__ float bf16_bits_to_f32(unsigned short h) {
  return __uint_as_float(((unsigned)h) << 16);
}
__device__ __forceinline__ float relu_f(float v) { return v > 0.f ? v : 0.f; }

// ---------------------------------------------------------------------------
// Kernel 1: split Omega (500x500 f32, row-major [k][n]) into transposed,
// zero-padded 512x512 bf16 hi/lo arrays Ot[n][k]. Layout makes the WMMA
// B-operand a pair of contiguous 16-byte loads per lane.
// ---------------------------------------------------------------------------
__global__ void omega_prep(const float* __restrict__ Omega,
                           unsigned short* __restrict__ ot_hi,
                           unsigned short* __restrict__ ot_lo) {
  int idx = blockIdx.x * blockDim.x + threadIdx.x;   // 0 .. 512*512-1
  int n = idx & (KPAD - 1);
  int k = idx >> 9;
  float v = 0.f;
  if (n < IN_DIM && k < IN_DIM) v = Omega[(size_t)k * IN_DIM + n];
  unsigned short hi = f32_to_bf16_rn(v);
  float lo = v - bf16_bits_to_f32(hi);
  ot_hi[(size_t)n * KPAD + k] = hi;
  ot_lo[(size_t)n * KPAD + k] = f32_to_bf16_rn(lo);
}

// ---------------------------------------------------------------------------
// Kernel 2: per-row O(D) statistics. One wave (32 lanes) per row.
// Writes base[i] (all non-dQd relu terms), l2[i] = d.alpha, rowabs -> out tmp.
// ---------------------------------------------------------------------------
__global__ __launch_bounds__(256) void row_stats(
    const float* __restrict__ x, const float* __restrict__ x_bw,
    const float* __restrict__ alpha, const float* __restrict__ beta,
    const int* __restrict__ sector_id, const int* __restrict__ mq_id,
    float* __restrict__ base, float* __restrict__ l2arr,
    float* __restrict__ rowabs) {
  int lane = threadIdx.x & 31;
  int w    = threadIdx.x >> 5;
  int row  = blockIdx.x * 8 + w;
  const float* xr = x + (size_t)row * IN_DIM;

  float sx = 0.f, sa = 0.f, nnz = 0.f, db = 0.f, da = 0.f;
  float sec[NBSECTOR], mqs[NBMQ];
#pragma unroll
  for (int s = 0; s < NBSECTOR; ++s) sec[s] = 0.f;
#pragma unroll
  for (int m = 0; m < NBMQ; ++m) mqs[m] = 0.f;

  for (int k = lane; k < IN_DIM; k += 32) {
    float xv = xr[k];
    float d  = xv - x_bw[k];
    sx  += xv;
    sa  += fabsf(d);
    nnz += (xv > 0.001f) ? 1.f : 0.f;
    db  += d * beta[k];
    da  += d * alpha[k];
    int sid = sector_id[k];
    int mid = mq_id[k];
#pragma unroll
    for (int s = 0; s < NBSECTOR; ++s) sec[s] += (sid == s) ? d : 0.f;
#pragma unroll
    for (int m = 0; m < NBMQ; ++m)     mqs[m] += (mid == m) ? d : 0.f;
  }

#pragma unroll
  for (int off = 16; off >= 1; off >>= 1) {
    sx  += __shfl_xor(sx,  off, 32);
    sa  += __shfl_xor(sa,  off, 32);
    nnz += __shfl_xor(nnz, off, 32);
    db  += __shfl_xor(db,  off, 32);
    da  += __shfl_xor(da,  off, 32);
#pragma unroll
    for (int s = 0; s < NBSECTOR; ++s) sec[s] += __shfl_xor(sec[s], off, 32);
#pragma unroll
    for (int m = 0; m < NBMQ; ++m)     mqs[m] += __shfl_xor(mqs[m], off, 32);
  }

  if (lane == 0) {
    float tot = relu_f(1.f - sx) + relu_f(sx - 1.f) + relu_f(sa - 0.05f);
#pragma unroll
    for (int s = 0; s < NBSECTOR; ++s) tot += relu_f(fabsf(sec[s]) - 0.1f);
#pragma unroll
    for (int m = 0; m < NBMQ; ++m)     tot += relu_f(fabsf(mqs[m]) - 0.1f);
    tot += relu_f(fabsf(db) - 0.1f);
    tot += relu_f(nnz - 70.f) + relu_f(69.f - nnz);
    base[row]   = tot;
    l2arr[row]  = da;
    rowabs[row] = sa;
  }
}

// ---------------------------------------------------------------------------
// Kernel 3: deterministic tree reduction of per-row |d| sums -> scalar.
// ---------------------------------------------------------------------------
__global__ void abs_reduce(const float* __restrict__ rowabs,
                           float* __restrict__ out_scalar) {
  __shared__ float sm[512];
  float s = 0.f;
  for (int i = threadIdx.x; i < BATCH; i += 512) s += rowabs[i];
  sm[threadIdx.x] = s;
  __syncthreads();
  for (int off = 256; off > 0; off >>= 1) {
    if (threadIdx.x < (unsigned)off) sm[threadIdx.x] += sm[threadIdx.x + off];
    __syncthreads();
  }
  if (threadIdx.x == 0) out_scalar[0] = sm[0];
}

// ---------------------------------------------------------------------------
// Kernel 4: dQd via split-bf16 WMMA + fusion.
//  Stage 1: block cooperatively stages d = x - x_bw for its 32 rows x 512 K
//           into LDS as bf16 hi/lo (zero-padded), split done ONCE per block.
//  Stage 2: K loop: A from LDS (ds_load_b128 x4), B (Omega panels, L2-hot)
//           from global (b128 x4 per tile), 3 WMMAs per N-tile (split f32).
//  Stage 3: y .* d row-reduce (d rebuilt from LDS hi+lo), wave shuffle +
//           LDS combine, then the full relu/tanh tail.
//  8 waves/block = 32 rows/block; wave = 4*rowblock + colgroup (128 cols).
// ---------------------------------------------------------------------------
__global__ __launch_bounds__(256) void dqd_wmma(
    const float* __restrict__ x, const float* __restrict__ x_bw,
    const unsigned short* __restrict__ ot_hi,
    const unsigned short* __restrict__ ot_lo,
    const float* __restrict__ base, const float* __restrict__ l2arr,
    const float* __restrict__ sum_abs, float* __restrict__ out) {
  __shared__ unsigned short ah_s[32 * KSTRIDE];
  __shared__ unsigned short al_s[32 * KSTRIDE];
  __shared__ float dq[8][16];

  int tid  = threadIdx.x;
  int wave = tid >> 5, lane = tid & 31;

  // ---- Stage 1: stage split-bf16 d into LDS (each wave: 4 rows) ----
  {
    int base_row = blockIdx.x * 32;
#pragma unroll
    for (int rr = 0; rr < 4; ++rr) {
      int row_local = wave * 4 + rr;
      const float* xr = x + (size_t)(base_row + row_local) * IN_DIM;
#pragma unroll
      for (int i = 0; i < 8; ++i) {
        int k = lane * 2 + i * 64;           // even; IN_DIM even -> pair valid
        float d0 = 0.f, d1 = 0.f;
        if (k < IN_DIM) {
          float2 xv = *(const float2*)(xr + k);
          d0 = xv.x - x_bw[k];
          d1 = xv.y - x_bw[k + 1];
        }
        unsigned short h0 = f32_to_bf16_rn(d0);
        unsigned short h1 = f32_to_bf16_rn(d1);
        unsigned short l0 = f32_to_bf16_rn(d0 - bf16_bits_to_f32(h0));
        unsigned short l1 = f32_to_bf16_rn(d1 - bf16_bits_to_f32(h1));
        unsigned off = (unsigned)row_local * KSTRIDE + k;
        *(unsigned*)(ah_s + off) = (unsigned)h0 | ((unsigned)h1 << 16);
        *(unsigned*)(al_s + off) = (unsigned)l0 | ((unsigned)l1 << 16);
      }
    }
  }
  __syncthreads();

  int lm = lane & 15, h = lane >> 4;
  int rb = wave >> 2, cg = wave & 3;
  int n0base = cg * 128;
  const unsigned short* a_hi = ah_s + (unsigned)(rb * 16 + lm) * KSTRIDE;
  const unsigned short* a_lo = al_s + (unsigned)(rb * 16 + lm) * KSTRIDE;

  v8f c[8];
#pragma unroll
  for (int t = 0; t < 8; ++t) c[t] = (v8f){0.f,0.f,0.f,0.f,0.f,0.f,0.f,0.f};

  // ---- Stage 2: K loop ----
  for (int kt = 0; kt < 16; ++kt) {
    int k0 = kt * 32;
    BF16Vec ah, al;
    ah.q[0] = *(const uint4*)(a_hi + k0 + 16 * h);       // A: K 16h..16h+7
    ah.q[1] = *(const uint4*)(a_hi + k0 + 16 * h + 8);   // A: K 16h+8..+7
    al.q[0] = *(const uint4*)(a_lo + k0 + 16 * h);
    al.q[1] = *(const uint4*)(a_lo + k0 + 16 * h + 8);
#pragma unroll
    for (int t = 0; t < 8; ++t) {
      size_t boff = (size_t)(n0base + t * 16 + lm) * KPAD + k0 + 16 * h;
      const uint4* ph = (const uint4*)(ot_hi + boff);
      const uint4* pl = (const uint4*)(ot_lo + boff);
      BF16Vec bh, bl;
      bh.q[0] = ph[0]; bh.q[1] = ph[1];
      bl.q[0] = pl[0]; bl.q[1] = pl[1];
      c[t] = __builtin_amdgcn_wmma_f32_16x16x32_bf16(false, ah.v, false, bh.v,
                                                     (short)0, c[t], false, false);
      c[t] = __builtin_amdgcn_wmma_f32_16x16x32_bf16(false, ah.v, false, bl.v,
                                                     (short)0, c[t], false, false);
      c[t] = __builtin_amdgcn_wmma_f32_16x16x32_bf16(false, al.v, false, bh.v,
                                                     (short)0, c[t], false, false);
    }
  }

  // ---- Stage 3: dQd partials = sum_n y[row][n] * d[row][n] ----
  float acc[8];
#pragma unroll
  for (int r = 0; r < 8; ++r) acc[r] = 0.f;
#pragma unroll
  for (int t = 0; t < 8; ++t) {
    int col = n0base + t * 16 + lm;          // padded cols give d == 0
#pragma unroll
    for (int r = 0; r < 8; ++r) {
      unsigned off = (unsigned)(rb * 16 + r + 8 * h) * KSTRIDE + col;
      float dv = bf16_bits_to_f32(ah_s[off]) + bf16_bits_to_f32(al_s[off]);
      acc[r] += c[t][r] * dv;
    }
  }
#pragma unroll
  for (int r = 0; r < 8; ++r) {
#pragma unroll
    for (int m = 8; m >= 1; m >>= 1) acc[r] += __shfl_xor(acc[r], m, 32);
  }
  if (lm == 0) {
#pragma unroll
    for (int r = 0; r < 8; ++r) dq[wave][8 * h + r] = acc[r];
  }
  __syncthreads();

  if (tid < 32) {
    int rb2 = tid >> 4, rr = tid & 15;
    float dQd = dq[rb2 * 4 + 0][rr] + dq[rb2 * 4 + 1][rr] +
                dq[rb2 * 4 + 2][rr] + dq[rb2 * 4 + 3][rr];
    int i = blockIdx.x * 32 + tid;
    float tot = base[i];
    float l_all = 0.5f * sum_abs[0];
    tot += relu_f(0.6f - l_all);
    tot += relu_f(dQd - 0.01f) + relu_f(0.0025f - dQd);
    tot += relu_f(100.f * dQd - 100.f * l2arr[i] - 1000.f);
    out[i] = relu_f(1.f - tanhf(tot * 0.01f));
  }
}

// ---------------------------------------------------------------------------
extern "C" void kernel_launch(void* const* d_in, const int* in_sizes, int n_in,
                              void* d_out, int out_size, void* d_ws, size_t ws_size,
                              hipStream_t stream) {
  (void)in_sizes; (void)n_in; (void)out_size; (void)ws_size;
  const float* x      = (const float*)d_in[0];
  const float* x_bw   = (const float*)d_in[1];
  const float* alpha  = (const float*)d_in[2];
  const float* beta   = (const float*)d_in[3];
  const float* Omega  = (const float*)d_in[4];
  const int* sector_id = (const int*)d_in[5];
  const int* mq_id     = (const int*)d_in[6];
  float* out = (float*)d_out;

  char* ws = (char*)d_ws;
  unsigned short* ot_hi = (unsigned short*)(ws);                 // 512 KB
  unsigned short* ot_lo = (unsigned short*)(ws + 512 * 1024);    // 512 KB
  float* base    = (float*)(ws + 1024 * 1024);                   // 512 KB
  float* l2arr   = (float*)(ws + 1536 * 1024);                   // 512 KB
  float* sum_abs = (float*)(ws + 2048 * 1024);                   // 4 B

  omega_prep<<<(KPAD * KPAD) / 256, 256, 0, stream>>>(Omega, ot_hi, ot_lo);
  // rowabs staged in d_out (overwritten by dqd_wmma afterwards)
  row_stats<<<BATCH / 8, 256, 0, stream>>>(x, x_bw, alpha, beta,
                                           sector_id, mq_id, base, l2arr, out);
  abs_reduce<<<1, 512, 0, stream>>>(out, sum_abs);
  dqd_wmma<<<BATCH / 32, 256, 0, stream>>>(x, x_bw, ot_hi, ot_lo,
                                           base, l2arr, sum_abs, out);
}
